// ODST_2860448219730
// MI455X (gfx1250) — matrix operand order repaired and checked by reading, not compiled, not run on hardware
//
#include <hip/hip_runtime.h>
#include <hip/hip_bf16.h>
#include <cstdint>
#include <cstddef>

typedef _Float16 v8h  __attribute__((ext_vector_type(8)));
typedef _Float16 v16h __attribute__((ext_vector_type(16)));
typedef float    v8f  __attribute__((ext_vector_type(8)));
typedef int      vi4  __attribute__((vector_size(16)));  // int4 vector for async builtin

#define AS1 __attribute__((address_space(1)))
#define AS3 __attribute__((address_space(3)))

#if defined(__has_builtin)
#if __has_builtin(__builtin_amdgcn_global_load_async_to_lds_b128)
#define HAVE_ASYNC_LDS 1
#endif
#endif

#define B_DIM 2048
#define I_DIM 1024
#define T_DIM 512
#define D_DIM 6
#define L_DIM 64
#define N_DIM (T_DIM * D_DIM)   /* 3072 */
#define TK 32                   /* K per WMMA step  */
#define LDA 40                  /* padded LDS stride in halves (80B) */

__device__ __forceinline__ void wait_async0() {
#if defined(__has_builtin) && __has_builtin(__builtin_amdgcn_s_wait_asynccnt)
  __builtin_amdgcn_s_wait_asynccnt(0);
#else
  asm volatile("s_wait_asynccnt 0x0" ::: "memory");
#endif
}

// ---------------------------------------------------------------------------
// Kernel 1: row-wise softmax over feature_selectors (3072 rows x 1024),
// output as _Float16 "gates" laid out (n = t*6+d) x (k = i), K-contiguous.
// ---------------------------------------------------------------------------
__global__ __launch_bounds__(256) void odst_softmax_f16(
    const float* __restrict__ fs, _Float16* __restrict__ gates) {
  __shared__ float red[256];
  const int row = blockIdx.x;
  const int tid = threadIdx.x;
  const float* src = fs + (size_t)row * I_DIM;
  _Float16* dst = gates + (size_t)row * I_DIM;

  float v[4];
  float mx = -3.4e38f;
#pragma unroll
  for (int j = 0; j < 4; ++j) { v[j] = src[tid + j * 256]; mx = fmaxf(mx, v[j]); }
  red[tid] = mx; __syncthreads();
  for (int s = 128; s > 0; s >>= 1) {
    if (tid < s) red[tid] = fmaxf(red[tid], red[tid + s]);
    __syncthreads();
  }
  mx = red[0];
  __syncthreads();

  float e[4]; float sum = 0.f;
#pragma unroll
  for (int j = 0; j < 4; ++j) { e[j] = __expf(v[j] - mx); sum += e[j]; }
  red[tid] = sum; __syncthreads();
  for (int s = 128; s > 0; s >>= 1) {
    if (tid < s) red[tid] += red[tid + s];
    __syncthreads();
  }
  const float inv = 1.0f / red[0];
#pragma unroll
  for (int j = 0; j < 4; ++j) dst[tid + j * 256] = (_Float16)(e[j] * inv);
}

// ---------------------------------------------------------------------------
// Kernel 2: convert x (f32) -> f16
// ---------------------------------------------------------------------------
__global__ __launch_bounds__(256) void odst_cvt_f16(
    const float* __restrict__ x, _Float16* __restrict__ xh, int n) {
  const int i = blockIdx.x * 256 + threadIdx.x;
  if (i < n) xh[i] = (_Float16)x[i];
}

// ---------------------------------------------------------------------------
// Kernel 3: WMMA GEMM  C[M=2048, N=3072] = A[M,K=1024] * B[N,K]^T
// A = x_f16 (row-major M x K); B = gates_f16 (row-major N x K).
// 128x128 workgroup tile, 8 waves (4 along M x 2 along N),
// wave sub-tile 32x64 = 2x4 of 16x16 WMMA tiles. Double-buffered LDS.
// Preferred path: async global->LDS loads (no staging VGPRs, ASYNCcnt).
// ---------------------------------------------------------------------------
__global__ __launch_bounds__(256) __attribute__((amdgpu_waves_per_eu(1)))
void odst_wmma_gemm(
    const _Float16* __restrict__ A, const _Float16* __restrict__ Bm,
    float* __restrict__ C) {
  __shared__ __align__(16) _Float16 sm[2 * 2 * 128 * LDA];  // 40960 B

  const int tid  = threadIdx.x;
  const int lane = tid & 31;
  const int wid  = tid >> 5;       // 0..7
  const int wm   = wid >> 1;       // 0..3 -> M offset 32*wm
  const int wn   = wid & 1;        // 0..1 -> N offset 64*wn
  const int g    = lane >> 4;      // lane group (0: lanes 0-15, 1: 16-31)
  const int l16  = lane & 15;

  const int m0 = blockIdx.y * 128;
  const int n0 = blockIdx.x * 128;

  const int lrow = tid >> 2;       // 0..63 (global->LDS staging row)
  const int lchk = tid & 3;        // 16B chunk within a 64B row

  v8f acc[2][4];
#pragma unroll
  for (int i = 0; i < 2; ++i)
#pragma unroll
    for (int j = 0; j < 4; ++j) acc[i][j] = {};

#ifdef HAVE_ASYNC_LDS
  // ---- async path: memory -> LDS without VGPR staging (ASYNCcnt) ----
  auto aload = [&](int ks, int buf) {
    const size_t k0 = (size_t)ks * TK;
    _Float16* sa = sm + (size_t)buf * (2 * 128 * LDA);
    _Float16* sb = sa + 128 * LDA;
#pragma unroll
    for (int r = 0; r < 2; ++r) {
      const int row = lrow + r * 64;
      __builtin_amdgcn_global_load_async_to_lds_b128(
          (AS1 vi4*)(A + (size_t)(m0 + row) * I_DIM + k0 + lchk * 8),
          (AS3 vi4*)(sa + row * LDA + lchk * 8), 0, 0);
      __builtin_amdgcn_global_load_async_to_lds_b128(
          (AS1 vi4*)(Bm + (size_t)(n0 + row) * I_DIM + k0 + lchk * 8),
          (AS3 vi4*)(sb + row * LDA + lchk * 8), 0, 0);
    }
  };
#else
  // ---- fallback: register staging with short live ranges ----
  uint4 ra[2], rb[2];
  auto gload = [&](int ks) {
    const size_t k0 = (size_t)ks * TK;
#pragma unroll
    for (int r = 0; r < 2; ++r) {
      const int row = lrow + r * 64;
      ra[r] = *(const uint4*)(A  + (size_t)(m0 + row) * I_DIM + k0 + lchk * 8);
      rb[r] = *(const uint4*)(Bm + (size_t)(n0 + row) * I_DIM + k0 + lchk * 8);
    }
  };
  auto sstore = [&](int buf) {
    _Float16* a = sm + (size_t)buf * (2 * 128 * LDA);
    _Float16* b = a + 128 * LDA;
#pragma unroll
    for (int r = 0; r < 2; ++r) {
      const int row = lrow + r * 64;
      *(uint4*)(a + row * LDA + lchk * 8) = ra[r];
      *(uint4*)(b + row * LDA + lchk * 8) = rb[r];
    }
  };
#endif

#ifdef HAVE_ASYNC_LDS
  aload(0, 0);
  wait_async0();
#else
  gload(0);
  sstore(0);
#endif
  __syncthreads();

  int buf = 0;
  const int KS = I_DIM / TK;  // 32
#pragma unroll 1
  for (int ks = 0; ks < KS; ++ks) {
#ifdef HAVE_ASYNC_LDS
    if (ks + 1 < KS) aload(ks + 1, buf ^ 1);   // async fill of other buffer
#else
    if (ks + 1 < KS) { gload(ks + 1); sstore(buf ^ 1); }  // short live range
#endif

    const _Float16* a = sm + (size_t)buf * (2 * 128 * LDA);
    const _Float16* b = a + 128 * LDA;

    // A fragment: 16x32 f16. Lane (g,m): halves 0-7 = K 8g..8g+7,
    // halves 8-15 = K 16+8g..16+8g+7 of row m.
    union { v16h v; v8h h[2]; } fa[2], fb[4];
#pragma unroll
    for (int mi = 0; mi < 2; ++mi) {
      const int row = wm * 32 + mi * 16 + l16;
      fa[mi].h[0] = *(const v8h*)(a + row * LDA + 8 * g);
      fa[mi].h[1] = *(const v8h*)(a + row * LDA + 16 + 8 * g);
    }
    // B fragment: 32x16 f16, LDS tile stored N-major; lane (g,n): halves
    // 0-7 = K 8g..8g+7, halves 8-15 = K 16+8g..16+8g+7 of column n.
#pragma unroll
    for (int ni = 0; ni < 4; ++ni) {
      const int col = wn * 64 + ni * 16 + l16;
      fb[ni].h[0] = *(const v8h*)(b + col * LDA + 8 * g);
      fb[ni].h[1] = *(const v8h*)(b + col * LDA + 16 + 8 * g);
    }

#pragma unroll
    for (int mi = 0; mi < 2; ++mi)
#pragma unroll
      for (int ni = 0; ni < 4; ++ni)
        acc[mi][ni] = __builtin_amdgcn_wmma_f32_16x16x32_f16(
            false, fa[mi].v, false, fb[ni].v, (short)0, acc[mi][ni],
            false, false);

#ifdef HAVE_ASYNC_LDS
    if (ks + 1 < KS) wait_async0();  // other buffer filled before barrier
#endif
    __syncthreads();
    buf ^= 1;
  }

  // C layout: VGPR r, lane (g, n=l16): element (M = r + 8g, N = n)
#pragma unroll
  for (int mi = 0; mi < 2; ++mi)
#pragma unroll
    for (int ni = 0; ni < 4; ++ni) {
      const int col = n0 + wn * 64 + ni * 16 + l16;
#pragma unroll
      for (int r = 0; r < 8; ++r) {
        const int row = m0 + wm * 32 + mi * 16 + 8 * g + r;
        C[(size_t)row * N_DIM + col] = acc[mi][ni][r];
      }
    }
}

// ---------------------------------------------------------------------------
// Kernel 4: epilogue. h = sigmoid(xp - thr); expand to 64 leaf probs via
// binary-tree product; dot with leaf_responses. One thread per (b, t).
// Block covers 128 consecutive t for one b; leaf_responses cached in LDS.
// ---------------------------------------------------------------------------
__global__ __launch_bounds__(128) void odst_epilogue(
    const float* __restrict__ xp, const float* __restrict__ thr,
    const float* __restrict__ lr, float* __restrict__ out) {
  __shared__ float s_lr[128 * 65];   // stride 65 to avoid bank conflicts
  __shared__ float s_th[128 * 6];
  const int tid = threadIdx.x;
  const int t0 = blockIdx.x * 128;
  const int b  = blockIdx.y;

  for (int j = tid; j < 128 * 64; j += 128)
    s_lr[(j >> 6) * 65 + (j & 63)] = lr[(size_t)t0 * 64 + j];
  for (int j = tid; j < 128 * 6; j += 128)
    s_th[j] = thr[(size_t)t0 * 6 + j];
  __syncthreads();

  const int t = t0 + tid;
  const float* xrow = xp + (size_t)b * N_DIM + (size_t)t * D_DIM;

  float p[L_DIM];
  p[0] = 1.f;
#pragma unroll
  for (int d = 0; d < D_DIM; ++d) {
    const float z = xrow[d] - s_th[tid * 6 + d];
    const float h = 1.0f / (1.0f + __expf(-z));
    const int n = 1 << d;
#pragma unroll
    for (int j = n - 1; j >= 0; --j) {
      p[j + n] = p[j] * h;          // bit d == 1
      p[j]     = p[j] * (1.0f - h); // bit d == 0
    }
  }
  float acc = 0.f;
#pragma unroll
  for (int l = 0; l < L_DIM; ++l) acc += p[l] * s_lr[tid * 65 + l];
  out[(size_t)b * T_DIM + t] = acc;
}

// ---------------------------------------------------------------------------
extern "C" void kernel_launch(void* const* d_in, const int* in_sizes, int n_in,
                              void* d_out, int out_size, void* d_ws,
                              size_t ws_size, hipStream_t stream) {
  const float* x   = (const float*)d_in[0];  // (B, I)
  const float* fs  = (const float*)d_in[1];  // (T, D, I)
  const float* thr = (const float*)d_in[2];  // (T, D)
  const float* lr  = (const float*)d_in[3];  // (T, L)
  float* out = (float*)d_out;                // (B, T)

  // workspace layout: gates f16 (6 MB) | x f16 (4 MB) | x_proj f32 (24 MB)
  char* ws = (char*)d_ws;
  _Float16* gates = (_Float16*)ws;
  _Float16* x16   = (_Float16*)(ws + (size_t)N_DIM * I_DIM * 2);
  float*    xpj   = (float*)(ws + (size_t)N_DIM * I_DIM * 2 +
                                  (size_t)B_DIM * I_DIM * 2);

  odst_softmax_f16<<<N_DIM, 256, 0, stream>>>(fs, gates);
  odst_cvt_f16<<<(B_DIM * I_DIM) / 256, 256, 0, stream>>>(x, x16,
                                                          B_DIM * I_DIM);
  odst_wmma_gemm<<<dim3(N_DIM / 128, B_DIM / 128), 256, 0, stream>>>(
      x16, gates, xpj);
  odst_epilogue<<<dim3(T_DIM / 128, B_DIM), 128, 0, stream>>>(xpj, thr, lr,
                                                              out);
}
